// NonAutoregressiveDecoder_48120813584451
// MI455X (gfx1250) — matrix operand order repaired and checked by reading, not compiled, not run on hardware
//
#include <hip/hip_runtime.h>
#include <stdint.h>

// Problem constants (from reference): B, E, D, L, N, T
#define B_ 32
#define E_ 32000
#define D_ 128
#define L_ 3
#define N_ 1000
#define T_ 50

typedef _Float16 v16h __attribute__((ext_vector_type(16)));
typedef _Float16 v8h  __attribute__((ext_vector_type(8)));
typedef float    v8f  __attribute__((ext_vector_type(8)));
typedef float    v4f  __attribute__((ext_vector_type(4)));

// XOR-swizzled LDS addressing (half-element index) for a [row][128] f16 tile:
// 16-half (32B) chunks are permuted by row so that lanes reading different rows
// at the same k-offset hit different bank groups (avoids 256B-stride conflicts).
__device__ __forceinline__ int swz(int row, int kh) {
  return row * D_ + ((((kh >> 4) ^ (row & 7)) << 4) | (kh & 15));
}

// SiLU with hardware reciprocal (v_exp_f32 + v_add + v_rcp_f32 + v_mul),
// avoiding the precise-division expansion (div_scale/div_fmas/div_fixup).
__device__ __forceinline__ float silu_f(float x) {
  return x * __builtin_amdgcn_rcpf(1.0f + __expf(-x));
}

// ---------------------------------------------------------------------------
// Kernel 1: fused 3-layer SiLU MLP (f16 WMMA, f32 accum) + Wout dot -> logits
// One wave (32 lanes) computes a 16-row x 128-feature tile.
// Block = 256 threads = 8 waves = 128 rows. Grid = B*E/128 = 8000.
// ---------------------------------------------------------------------------
__global__ __launch_bounds__(256) void mlp_logits_kernel(
    const float* __restrict__ edge_attr,   // (B*E, 128) fp32
    const float* __restrict__ Ws,          // (3, 128, 128) fp32, [l][k][n]
    const float* __restrict__ bs,          // (3, 128) fp32
    const float* __restrict__ Wout,        // (128,) fp32
    const float* __restrict__ bout,        // (1,) fp32
    float* __restrict__ logits)            // (B*E,) fp32
{
  __shared__ _Float16 wlds[D_ * D_];        // 32KB: layer weights, transposed [n][k], swizzled
  __shared__ _Float16 stage[8 * 16 * D_];   // 32KB: per-wave 16x128 activation staging, swizzled

  const int tid  = threadIdx.x;
  const int lane = tid & 31;
  const int wave = tid >> 5;
  const int hi   = lane >> 4;     // half-wave select (A/B fragment K-interleave)
  const int lm   = lane & 15;
  const int base_row = blockIdx.x * 128 + wave * 16;

  _Float16* mystage = stage + wave * 16 * D_;

  // Preload Wout (one value per lane-column, all 8 n-tiles) before the loop.
  float woutv[8];
#pragma unroll
  for (int nt = 0; nt < 8; ++nt) woutv[nt] = Wout[nt * 16 + lm];

  v16h afrag[4];                  // 16x128 f16 A tile (4 k-steps of 32)
  float partial[8];
#pragma unroll
  for (int v = 0; v < 8; ++v) partial[v] = 0.0f;

#pragma unroll
  for (int l = 0; l < L_; ++l) {
    __syncthreads();  // all waves done reading previous layer's weights
    {
      // Cooperative load: Wt[n][k] = Ws[l][k][n], fp32 -> f16, swizzled
      const float* wsrc = Ws + (size_t)l * D_ * D_;
      for (int i = tid; i < D_ * D_; i += 256) {
        int n = i >> 7;
        int k = i & 127;
        wlds[swz(n, k)] = (_Float16)wsrc[(size_t)k * D_ + n];
      }
    }

    // Preload this layer's bias values (per lane-column, all 8 n-tiles) so the
    // global loads complete before the WMMA steady state.
    float biasv[8];
#pragma unroll
    for (int nt = 0; nt < 8; ++nt) biasv[nt] = bs[l * D_ + nt * 16 + lm];

    __syncthreads();

    // ---- Load A fragments (ISA 16-bit A layout: lane<16 K{0-7,16-23}, lane>=16 K{8-15,24-31}) ----
    if (l == 0) {
#pragma unroll
      for (int kt = 0; kt < 4; ++kt) {
        const float* rp = edge_attr + (size_t)(base_row + lm) * D_ + kt * 32 + hi * 8;
        const v4f* rp4 = (const v4f*)rp;
        v4f x0 = rp4[0], x1 = rp4[1];   // K offsets +0..7
        v4f x2 = rp4[4], x3 = rp4[5];   // K offsets +16..23
        v16h a;
#pragma unroll
        for (int j = 0; j < 4; ++j) {
          a[j]      = (_Float16)x0[j];
          a[4 + j]  = (_Float16)x1[j];
          a[8 + j]  = (_Float16)x2[j];
          a[12 + j] = (_Float16)x3[j];
        }
        afrag[kt] = a;
      }
    } else {
#pragma unroll
      for (int kt = 0; kt < 4; ++kt) {
        int kh0 = kt * 32 + hi * 8;
        const v8h* p0 = (const v8h*)&mystage[swz(lm, kh0)];
        const v8h* p1 = (const v8h*)&mystage[swz(lm, kh0 + 16)];
        v8h lo = p0[0];
        v8h hh = p1[0];
        afrag[kt] = __builtin_shufflevector(lo, hh,
            0, 1, 2, 3, 4, 5, 6, 7, 8, 9, 10, 11, 12, 13, 14, 15);
      }
    }

    // ---- 8 n-tiles x 4 k-steps of WMMA ----
#pragma unroll
    for (int nt = 0; nt < 8; ++nt) {
      // Prefetch all 4 B fragments for this n-tile (8x ds_load_b128 batched,
      // single dscnt wait), then run 4 back-to-back WMMAs chained on C.
      v16h bfr[4];
#pragma unroll
      for (int kt = 0; kt < 4; ++kt) {
        const v8h* wp = (const v8h*)&wlds[swz(nt * 16 + lm, kt * 32 + hi * 8)];
        const v8h* wq = (const v8h*)&wlds[swz(nt * 16 + lm, kt * 32 + hi * 8 + 16)];
        v8h b0 = wp[0];
        v8h b1 = wq[0];
        bfr[kt] = __builtin_shufflevector(b0, b1,
            0, 1, 2, 3, 4, 5, 6, 7, 8, 9, 10, 11, 12, 13, 14, 15);
      }

      v8f acc = {};
#pragma unroll
      for (int kt = 0; kt < 4; ++kt) {
        acc = __builtin_amdgcn_wmma_f32_16x16x32_f16(
            /*neg_a=*/false, afrag[kt], /*neg_b=*/false, bfr[kt],
            /*c_mod=*/(short)0, acc, /*reuse_a=*/false, /*reuse_b=*/false);
      }

      // C/D layout: VGPR v -> row (v + hi*8), lane -> column (nt*16 + lm)
      float bias = biasv[nt];
      if (l < L_ - 1) {
        int n = nt * 16 + lm;
#pragma unroll
        for (int v = 0; v < 8; ++v) {
          float s = silu_f(acc[v] + bias);
          int m = v + hi * 8;
          mystage[swz(m, n)] = (_Float16)s;   // f16 staging for next layer's A
        }
      } else {
        float wv = woutv[nt];
#pragma unroll
        for (int v = 0; v < 8; ++v) {
          float s = silu_f(acc[v] + bias);
          partial[v] += s * wv;               // fold Wout dot into epilogue
        }
      }
    }
  }

  // Reduce partial dot products across each 16-lane half (wave32 shuffles)
  float b0 = bout[0];
#pragma unroll
  for (int v = 0; v < 8; ++v) {
    float r = partial[v];
    r += __shfl_xor(r, 1, 32);
    r += __shfl_xor(r, 2, 32);
    r += __shfl_xor(r, 4, 32);
    r += __shfl_xor(r, 8, 32);
    if (lm == 0) {
      int m = v + hi * 8;                     // lane 0 -> row v, lane 16 -> row v+8
      logits[base_row + m] = r + b0;
    }
  }
}

// ---------------------------------------------------------------------------
// Kernel 2: winner[t,b,n] = -1
// ---------------------------------------------------------------------------
__global__ void init_winner_kernel(int* __restrict__ winner, long long total) {
  long long i = (long long)blockIdx.x * blockDim.x + threadIdx.x;
  if (i < total) winner[i] = -1;
}

// ---------------------------------------------------------------------------
// Kernel 3: deterministic scatter. For every edge e of batch b, for every t
// with actions[t,b] == src(e): winner[t,b,dst(e)] = max(winner, e).
// Highest edge index wins ties (deterministic across replays).
// ---------------------------------------------------------------------------
__global__ void scatter_winner_kernel(
    const long long* __restrict__ edge_index,  // (B, 2, E) int64
    const long long* __restrict__ actions,     // (T, B) int64
    int* __restrict__ winner)                  // (T, B, N) int32
{
  __shared__ int acts[T_];
  const int b = blockIdx.x;
  const int tid = threadIdx.x;
  if (tid < T_) acts[tid] = (int)actions[(size_t)tid * B_ + b];
  __syncthreads();

  const long long* src = edge_index + (size_t)b * 2 * E_;
  const long long* dst = src + E_;
  for (int e = tid; e < E_; e += blockDim.x) {
    int s = (int)src[e];
    int d = (int)dst[e];
#pragma unroll 1
    for (int t = 0; t < T_; ++t) {
      if (acts[t] == s) {
        atomicMax(&winner[((size_t)t * B_ + b) * N_ + d], e);
      }
    }
  }
}

// ---------------------------------------------------------------------------
// Kernel 4: out[t,b,n] = mask==0 ? -inf : (winner>=0 ? logits[b, winner] : -1e9)
// ---------------------------------------------------------------------------
__global__ void finalize_kernel(
    const int* __restrict__ winner,
    const float* __restrict__ logits,
    const int* __restrict__ action_mask,
    float* __restrict__ out,
    long long total)
{
  long long i = (long long)blockIdx.x * blockDim.x + threadIdx.x;
  if (i >= total) return;
  float r;
  if (action_mask[i] == 0) {
    r = -__builtin_inff();
  } else {
    int w = winner[i];
    if (w >= 0) {
      int b = (int)((i / N_) % B_);
      r = logits[(size_t)b * E_ + w];
    } else {
      r = -1.0e9f;
    }
  }
  out[i] = r;
}

// ---------------------------------------------------------------------------
extern "C" void kernel_launch(void* const* d_in, const int* in_sizes, int n_in,
                              void* d_out, int out_size, void* d_ws, size_t ws_size,
                              hipStream_t stream) {
  (void)in_sizes; (void)n_in; (void)out_size; (void)ws_size;

  const float*     edge_attr   = (const float*)d_in[0];
  const float*     Ws          = (const float*)d_in[1];
  const float*     bs          = (const float*)d_in[2];
  const float*     Wout        = (const float*)d_in[3];
  const float*     bout        = (const float*)d_in[4];
  const long long* edge_index  = (const long long*)d_in[5];
  const long long* actions     = (const long long*)d_in[6];
  const int*       action_mask = (const int*)d_in[7];
  float*           out         = (float*)d_out;

  // Workspace layout: logits (B*E f32, 4MB) | winner (T*B*N i32, 6.4MB)
  float* logits = (float*)d_ws;
  int*   winner = (int*)((char*)d_ws + (size_t)B_ * E_ * sizeof(float));

  const int rows = B_ * E_;                       // 1,024,000 (divisible by 128)
  mlp_logits_kernel<<<rows / 128, 256, 0, stream>>>(edge_attr, Ws, bs, Wout, bout, logits);

  const long long total = (long long)T_ * B_ * N_;  // 1,600,000
  init_winner_kernel<<<(int)((total + 255) / 256), 256, 0, stream>>>(winner, total);
  scatter_winner_kernel<<<B_, 256, 0, stream>>>(edge_index, actions, winner);
  finalize_kernel<<<(int)((total + 255) / 256), 256, 0, stream>>>(winner, logits, action_mask, out, total);
}
